// Qwen3_5MoeGatedSDPAAttention_33371895890025
// MI455X (gfx1250) — compile-verified, matrix operand order checked by reading, not run on hardware
//
#include <hip/hip_runtime.h>
#include <stdint.h>
#include <stddef.h>

// ---------------- problem constants ----------------
static constexpr int Bc   = 2;
static constexpr int Sc   = 2048;
static constexpr int HIDc = 2048;
static constexpr int Hc   = 16;
static constexpr int KVc  = 4;
static constexpr int Dc   = 128;
static constexpr int HDc  = Hc * Dc;       // 2048
static constexpr float EPSc   = 1e-6f;
static constexpr float SCALEc = 0.08838834764831845f; // D^-0.5

typedef __bf16 bf16;
typedef __attribute__((ext_vector_type(16))) __bf16 v16bf;
typedef __attribute__((ext_vector_type(8)))  float  v8f;

__device__ __forceinline__ v8f vzero() {
  v8f z = {0.f, 0.f, 0.f, 0.f, 0.f, 0.f, 0.f, 0.f};
  return z;
}

// round-to-nearest-even f32 -> bf16 (bit-exact, no reliance on native __bf16 cvt)
__device__ __forceinline__ bf16 f2bf(float f) {
  union { float f; uint32_t u; } x; x.f = f;
  uint32_t r = (x.u + 0x7FFFu + ((x.u >> 16) & 1u)) >> 16;
  uint16_t s = (uint16_t)r;
  bf16 b;
  __builtin_memcpy(&b, &s, 2);
  return b;
}

// ---------------- f32 -> bf16 convert ----------------
__global__ void k_f32_to_bf16(const float* __restrict__ in, bf16* __restrict__ out, int n) {
  int i = blockIdx.x * blockDim.x + threadIdx.x;
  if (i < n) out[i] = f2bf(in[i]);
}

// ---------------- generic bf16 GEMM: C[M,N] = A[M,K] * Bw[N,K]^T ----------------
// block = 256 threads (8 waves); block tile 128M x 128N; wave tile 16M x 128N.
__global__ void __launch_bounds__(256)
k_gemm_bf16(const bf16* __restrict__ A, const bf16* __restrict__ Bw,
            float* __restrict__ C, int M, int N, int K) {
  const int lane = threadIdx.x & 31;
  const int wave = threadIdx.x >> 5;
  const int m0 = blockIdx.x * 128 + wave * 16;
  const int n0 = blockIdx.y * 128;
  const int r16 = lane & 15;
  const int kh  = (lane >> 4) * 16;        // K sub-offset per lane half

  v8f acc[8];
#pragma unroll
  for (int c = 0; c < 8; ++c) acc[c] = vzero();

  const bf16* arow = A + (size_t)(m0 + r16) * K;
  for (int kk = 0; kk < K; kk += 32) {
    v16bf a = *reinterpret_cast<const v16bf*>(arow + kk + kh);
#pragma unroll
    for (int c = 0; c < 8; ++c) {
      const bf16* brow = Bw + (size_t)(n0 + c * 16 + r16) * K;
      v16bf b = *reinterpret_cast<const v16bf*>(brow + kk + kh);
      acc[c] = __builtin_amdgcn_wmma_f32_16x16x32_bf16(false, a, false, b,
                                                       (short)0, acc[c], false, false);
    }
  }

  // C layout: VGPR r -> row m0 + r + 8*(lane>=16); lane&15 -> column within 16-tile
  const int rbase = (lane >> 4) * 8;
#pragma unroll
  for (int c = 0; c < 8; ++c) {
#pragma unroll
    for (int r = 0; r < 8; ++r) {
      C[(size_t)(m0 + rbase + r) * N + (n0 + c * 16 + r16)] = acc[c][r];
    }
  }
}

// ---------------- RMS norm + RoPE (+ optional scale), writes (B, NH, S, D) bf16 ----------------
// one wave per (b, s, head) row of D=128; 4 waves / block.
__global__ void __launch_bounds__(128)
k_rmsnorm_rope(const float* __restrict__ src, const float* __restrict__ cosT,
               const float* __restrict__ sinT, const float* __restrict__ w,
               bf16* __restrict__ dst, int NH, int groupStride, int headStride, float scale) {
  const int lane = threadIdx.x & 31;
  const int wave = threadIdx.x >> 5;
  int rowid = blockIdx.x * 4 + wave;                 // 0 .. B*S*NH-1
  int head = rowid % NH;
  int s    = (rowid / NH) % Sc;
  int b    = rowid / (NH * Sc);

  const float* sp = src + (size_t)(b * Sc + s) * groupStride + (size_t)head * headStride;
  float v0 = sp[lane], v1 = sp[lane + 32], v2 = sp[lane + 64], v3 = sp[lane + 96];

  float ss = v0 * v0 + v1 * v1 + v2 * v2 + v3 * v3;
#pragma unroll
  for (int m = 1; m < 32; m <<= 1) ss += __shfl_xor(ss, m, 32);
  float inv = rsqrtf(ss * (1.0f / (float)Dc) + EPSc);

  v0 *= inv * (1.0f + w[lane]);
  v1 *= inv * (1.0f + w[lane + 32]);
  v2 *= inv * (1.0f + w[lane + 64]);
  v3 *= inv * (1.0f + w[lane + 96]);

  const float* cp = cosT + (size_t)(b * Sc + s) * Dc;
  const float* pp = sinT + (size_t)(b * Sc + s) * Dc;
  // rotate_half: d<64 -> -x[d+64]; d>=64 -> +x[d-64]; lane-local with our 4-way split.
  float o0 = (v0 * cp[lane]      - v2 * pp[lane])      * scale;
  float o1 = (v1 * cp[lane + 32] - v3 * pp[lane + 32]) * scale;
  float o2 = (v2 * cp[lane + 64] + v0 * pp[lane + 64]) * scale;
  float o3 = (v3 * cp[lane + 96] + v1 * pp[lane + 96]) * scale;

  bf16* dp = dst + ((size_t)(b * NH + head) * Sc + s) * Dc;
  dp[lane]      = f2bf(o0);
  dp[lane + 32] = f2bf(o1);
  dp[lane + 64] = f2bf(o2);
  dp[lane + 96] = f2bf(o3);
}

// ---------------- V: (B,S,KV,D) f32 -> (B,KV,S,D) bf16 ----------------
__global__ void k_v_reshape(const float* __restrict__ Vp, bf16* __restrict__ Vb, int n) {
  int i = blockIdx.x * blockDim.x + threadIdx.x;
  if (i >= n) return;
  int d  = i % Dc;
  int t  = i / Dc;
  int kv = t % KVc; t /= KVc;
  int s  = t % Sc;
  int b  = t / Sc;
  Vb[(((size_t)b * KVc + kv) * Sc + s) * Dc + d] = f2bf(Vp[i]);
}

// ---------------- flash attention + sigmoid gate ----------------
// grid: (S/64, H, B); block = 128 threads = 4 waves; wave owns 16 queries x D=128.
static constexpr int VTS = 48;   // LDS stride (elems) for transposed V tile; 96B, 32B-aligned rows
__global__ void __launch_bounds__(128)
k_attn(const bf16* __restrict__ Qb, const bf16* __restrict__ Kb, const bf16* __restrict__ Vb,
       const float* __restrict__ QG, bf16* __restrict__ AG) {
  __shared__ __align__(64) bf16 VT[Dc * VTS];     // [dim][key] transposed 32-key V tile
  __shared__ __align__(64) bf16 PS[4][16 * 32];   // per-wave P staging (A-operand relayout)

  const int lane = threadIdx.x & 31;
  const int wave = threadIdx.x >> 5;
  const int b  = blockIdx.z;
  const int h  = blockIdx.y;
  const int kv = h / (Hc / KVc);
  const int q0w  = blockIdx.x * 64 + wave * 16;   // this wave's first query
  const int jmax = blockIdx.x * 64 + 64;          // causal bound for the block (uniform)

  const bf16* qp = Qb + (((size_t)b * Hc  + h ) * Sc + q0w) * Dc;
  const bf16* kp = Kb +  ((size_t)b * KVc + kv) * Sc * Dc;
  const bf16* vp = Vb +  ((size_t)b * KVc + kv) * Sc * Dc;

  const int r16   = lane & 15;
  const int kh    = (lane >> 4) * 16;
  const int rbase = (lane >> 4) * 8;

  // preload Q tile as 4 A-operand chunks (16 x 32 each)
  v16bf aq[4];
#pragma unroll
  for (int ch = 0; ch < 4; ++ch)
    aq[ch] = *reinterpret_cast<const v16bf*>(qp + (size_t)r16 * Dc + ch * 32 + kh);

  v8f acc[8];
#pragma unroll
  for (int c = 0; c < 8; ++c) acc[c] = vzero();
  float mrow[8], lrow[8];
#pragma unroll
  for (int r = 0; r < 8; ++r) { mrow[r] = -3.0e38f; lrow[r] = 0.0f; }

  for (int j = 0; j < jmax; j += 32) {
    // cooperative transpose of 32-key V tile into LDS: thread -> (key, 32-dim slab)
    {
      int key = threadIdx.x & 31;
      int db  = (threadIdx.x >> 5) * 32;
      const bf16* vrow = vp + (size_t)(j + key) * Dc + db;
      v16bf x0 = *reinterpret_cast<const v16bf*>(vrow);
      v16bf x1 = *reinterpret_cast<const v16bf*>(vrow + 16);
#pragma unroll
      for (int e = 0; e < 16; ++e) {
        VT[(db + e)      * VTS + key] = x0[e];
        VT[(db + 16 + e) * VTS + key] = x1[e];
      }
    }
    __syncthreads();

    // scores: 16 queries x 32 keys, accumulate over D in 4 chunks of 32
    v8f s0 = vzero(), s1 = vzero();
#pragma unroll
    for (int ch = 0; ch < 4; ++ch) {
      v16bf b0 = *reinterpret_cast<const v16bf*>(kp + (size_t)(j + r16)      * Dc + ch * 32 + kh);
      v16bf b1 = *reinterpret_cast<const v16bf*>(kp + (size_t)(j + 16 + r16) * Dc + ch * 32 + kh);
      s0 = __builtin_amdgcn_wmma_f32_16x16x32_bf16(false, aq[ch], false, b0, (short)0, s0, false, false);
      s1 = __builtin_amdgcn_wmma_f32_16x16x32_bf16(false, aq[ch], false, b1, (short)0, s1, false, false);
    }

    // online softmax: rows live at (reg r, half); 16-lane xor reductions per half
#pragma unroll
    for (int r = 0; r < 8; ++r) {
      int qrow = q0w + rbase + r;
      float x0 = (j + r16      <= qrow) ? s0[r] : -3.0e38f;   // 1/sqrt(D) folded into Q
      float x1 = (j + 16 + r16 <= qrow) ? s1[r] : -3.0e38f;
      float mx = fmaxf(x0, x1);
      mx = fmaxf(mx, __shfl_xor(mx, 1, 32));
      mx = fmaxf(mx, __shfl_xor(mx, 2, 32));
      mx = fmaxf(mx, __shfl_xor(mx, 4, 32));
      mx = fmaxf(mx, __shfl_xor(mx, 8, 32));
      float mn = fmaxf(mrow[r], mx);
      float p0 = 0.f, p1 = 0.f, alpha = 1.f;
      if (mn > -1.0e37f) {
        p0 = __expf(x0 - mn);
        p1 = __expf(x1 - mn);
        alpha = __expf(mrow[r] - mn);
      }
      float ps = p0 + p1;
      ps += __shfl_xor(ps, 1, 32);
      ps += __shfl_xor(ps, 2, 32);
      ps += __shfl_xor(ps, 4, 32);
      ps += __shfl_xor(ps, 8, 32);
      lrow[r] = lrow[r] * alpha + ps;
      mrow[r] = mn;
#pragma unroll
      for (int c = 0; c < 8; ++c) acc[c][r] *= alpha;
      int prow = rbase + r;
      PS[wave][prow * 32 + r16]      = f2bf(p0);
      PS[wave][prow * 32 + 16 + r16] = f2bf(p1);
    }

    // O += P (16x32) x V (32x128): A from per-wave LDS stage, B from transposed V tile
    v16bf ap = *reinterpret_cast<const v16bf*>(&PS[wave][r16 * 32 + kh]);
#pragma unroll
    for (int c = 0; c < 8; ++c) {
      v16bf bv = *reinterpret_cast<const v16bf*>(&VT[(c * 16 + r16) * VTS + kh]);
      acc[c] = __builtin_amdgcn_wmma_f32_16x16x32_bf16(false, ap, false, bv, (short)0, acc[c], false, false);
    }
    __syncthreads();
  }

  // epilogue: normalize, sigmoid-gate (gate = QG[..., D:2D]), write bf16 (B,S,H*D)
#pragma unroll
  for (int r = 0; r < 8; ++r) {
    int qrow = q0w + rbase + r;
    float invl = 1.0f / fmaxf(lrow[r], 1e-20f);
    const float* gp = QG + ((size_t)(b * Sc + qrow) * Hc + h) * (2 * Dc) + Dc;
    bf16* op = AG + (size_t)(b * Sc + qrow) * HDc + h * Dc;
#pragma unroll
    for (int c = 0; c < 8; ++c) {
      int dim = c * 16 + r16;
      float g = gp[dim];
      float sig = 1.0f / (1.0f + __expf(-g));
      op[dim] = f2bf(acc[c][r] * invl * sig);
    }
  }
}

// ---------------- host launcher ----------------
extern "C" void kernel_launch(void* const* d_in, const int* in_sizes, int n_in,
                              void* d_out, int out_size, void* d_ws, size_t ws_size,
                              hipStream_t stream) {
  (void)in_sizes; (void)n_in; (void)out_size; (void)ws_size;
  const float* hs   = (const float*)d_in[0];
  const float* cosT = (const float*)d_in[1];
  const float* sinT = (const float*)d_in[2];
  const float* Wq   = (const float*)d_in[3];
  const float* Wk   = (const float*)d_in[4];
  const float* Wv   = (const float*)d_in[5];
  const float* Wo   = (const float*)d_in[6];
  const float* qw   = (const float*)d_in[7];
  const float* kw   = (const float*)d_in[8];
  float* out = (float*)d_out;

  char* ws = (char*)d_ws;
  size_t off = 0;
  auto alloc = [&](size_t bytes) -> void* {
    void* p = ws + off;
    off = (off + bytes + 255) & ~(size_t)255;
    return p;
  };

  const int nHS = Bc * Sc * HIDc;          // 8,388,608
  const int nWq = 2 * Hc * Dc * HIDc;      // 8,388,608
  const int nWk = KVc * Dc * HIDc;         // 1,048,576
  const int nWo = HIDc * HDc;              // 4,194,304
  const int nQG = Bc * Sc * 2 * Hc * Dc;   // 16,777,216
  const int nKP = Bc * Sc * KVc * Dc;      // 2,097,152
  const int nQ  = Bc * Hc * Sc * Dc;       // 8,388,608
  const int nKV = Bc * KVc * Sc * Dc;      // 2,097,152
  const int nAG = Bc * Sc * HDc;           // 8,388,608

  bf16* hsb = (bf16*)alloc((size_t)nHS * 2);
  bf16* wqb = (bf16*)alloc((size_t)nWq * 2);
  bf16* wkb = (bf16*)alloc((size_t)nWk * 2);
  bf16* wvb = (bf16*)alloc((size_t)nWk * 2);
  bf16* wob = (bf16*)alloc((size_t)nWo * 2);
  float* QG = (float*)alloc((size_t)nQG * 4);
  float* Kp = (float*)alloc((size_t)nKP * 4);
  float* Vp = (float*)alloc((size_t)nKP * 4);
  bf16* Qb  = (bf16*)alloc((size_t)nQ  * 2);
  bf16* Kb  = (bf16*)alloc((size_t)nKV * 2);
  bf16* Vb  = (bf16*)alloc((size_t)nKV * 2);
  bf16* AG  = (bf16*)alloc((size_t)nAG * 2);

  // 1) convert activations + weights to bf16
  k_f32_to_bf16<<<(nHS + 255) / 256, 256, 0, stream>>>(hs, hsb, nHS);
  k_f32_to_bf16<<<(nWq + 255) / 256, 256, 0, stream>>>(Wq, wqb, nWq);
  k_f32_to_bf16<<<(nWk + 255) / 256, 256, 0, stream>>>(Wk, wkb, nWk);
  k_f32_to_bf16<<<(nWk + 255) / 256, 256, 0, stream>>>(Wv, wvb, nWk);
  k_f32_to_bf16<<<(nWo + 255) / 256, 256, 0, stream>>>(Wo, wob, nWo);

  const int M = Bc * Sc;  // 4096
  // 2) projections
  k_gemm_bf16<<<dim3(M / 128, (2 * Hc * Dc) / 128), 256, 0, stream>>>(hsb, wqb, QG, M, 2 * Hc * Dc, HIDc);
  k_gemm_bf16<<<dim3(M / 128, (KVc * Dc) / 128),   256, 0, stream>>>(hsb, wkb, Kp, M, KVc * Dc, HIDc);
  k_gemm_bf16<<<dim3(M / 128, (KVc * Dc) / 128),   256, 0, stream>>>(hsb, wvb, Vp, M, KVc * Dc, HIDc);

  // 3) RMS norm + RoPE (Q gets 1/sqrt(D) folded in), V re-layout
  k_rmsnorm_rope<<<(Bc * Sc * Hc) / 4, 128, 0, stream>>>(QG, cosT, sinT, qw, Qb,
                                                         Hc, Hc * 2 * Dc, 2 * Dc, SCALEc);
  k_rmsnorm_rope<<<(Bc * Sc * KVc) / 4, 128, 0, stream>>>(Kp, cosT, sinT, kw, Kb,
                                                          KVc, KVc * Dc, Dc, 1.0f);
  k_v_reshape<<<(nKP + 255) / 256, 256, 0, stream>>>(Vp, Vb, nKP);

  // 4) flash attention + gate
  k_attn<<<dim3(Sc / 64, Hc, Bc), 128, 0, stream>>>(Qb, Kb, Vb, QG, AG);

  // 5) output projection -> f32 out
  k_gemm_bf16<<<dim3(M / 128, HIDc / 128), 256, 0, stream>>>(AG, wob, out, M, HIDc, HIDc);
}
